// PointNet2ClsModel_33079838113809
// MI455X (gfx1250) — matrix operand order repaired and checked by reading, not compiled, not run on hardware
//
#include <hip/hip_runtime.h>

typedef __attribute__((ext_vector_type(16))) _Float16 v16h;
typedef __attribute__((ext_vector_type(8)))  float    v8f;

union V16H { v16h h; uint4 q[2]; _Float16 e[16]; };

#define INVS 0.99999500003749971874f   // 1/sqrt(1+1e-5)

// ---------------- WMMA tile helper: A,B both f16 (LDS or global), 16B-aligned ----------------
// A layout: lane l -> row M=l&15, halves {kbA..kbA+7, kbA+16..kbA+23}, kbA=(l>=16)?8:0
// B layout: lane l -> col N=l&15, halves {kbB/2..kbB/2+7, kbB/2+16..kbB/2+23}, kbB=(l>=16)?16:0
__device__ __forceinline__ v8f tile_h(const _Float16* Arow, const _Float16* Brow,
                                      int Kpad, int kbA, int kbB) {
  v8f acc = {};
  for (int k0 = 0; k0 < Kpad; k0 += 32) {
    V16H a, b;
    a.q[0] = *(const uint4*)(Arow + k0 + kbA);
    a.q[1] = *(const uint4*)(Arow + k0 + kbA + 16);
    b.q[0] = *(const uint4*)(Brow + k0 + (kbB >> 1));
    b.q[1] = *(const uint4*)(Brow + k0 + (kbB >> 1) + 16);
    acc = __builtin_amdgcn_wmma_f32_16x16x32_f16(false, a.h, false, b.h, (short)0, acc, false, false);
  }
  return acc;
}

// One MLP layer: D(f16 LDS) = relu(bn(A(f16 LDS) x Wh^T)), block = 256 threads (8 waves).
// Wh: pre-converted f16 weights, row stride = Kpad (zero padded).
template<int ROWS>
__device__ __forceinline__ void layer_lds(const _Float16* A, int lda, int Kpad,
                                          const _Float16* Wh, const float* gam, const float* bet,
                                          _Float16* D, int ldd, int Cout) {
  const int lane = threadIdx.x & 31;
  const int wv   = threadIdx.x >> 5;
  const int lm   = lane & 15;
  const int hi   = lane >> 4;
  const int kbA  = hi ? 8 : 0;
  const int kbB  = hi ? 16 : 0;
  const int mt_n = ROWS / 16, nt_n = Cout / 16;
  for (int t = wv; t < mt_n * nt_n; t += 8) {
    int mt = t % mt_n, nt = t / mt_n;
    int ncol = nt * 16 + lm;
    v8f acc = tile_h(A + (mt * 16 + lm) * lda, Wh + ncol * Kpad, Kpad, kbA, kbB);
    float sc = gam[ncol] * INVS, bi = bet[ncol];
    int rb = mt * 16 + (hi ? 8 : 0);
#pragma unroll
    for (int r = 0; r < 8; ++r) {
      float v = acc[r] * sc + bi;
      D[(rb + r) * ldd + ncol] = (_Float16)(v > 0.f ? v : 0.f);
    }
  }
}

// ---------------- Weight conversion: f32 (Cout x Kreal) -> f16 (Cout x Kpad, zero padded) ----
__global__ void cvtw_kernel(const float* __restrict__ src, _Float16* __restrict__ dst,
                            int Cout, int Kreal, int Kpad) {
  int i = blockIdx.x * 256 + threadIdx.x;
  if (i < Cout * Kpad) {
    int r = i / Kpad, c = i % Kpad;
    dst[i] = (c < Kreal) ? (_Float16)src[r * Kreal + c] : (_Float16)0.f;
  }
}

// ---------------- FPS ----------------
template<int N, int NP, int ITER>
__global__ void fps_kernel(const float* __restrict__ xyz, int* __restrict__ cidx,
                           float* __restrict__ centers) {
  const int b = blockIdx.x, tid = threadIdx.x;
  __shared__ float sval[256];
  __shared__ int   sidx[256];
  __shared__ int   slast;
  float dm[ITER];
#pragma unroll
  for (int i = 0; i < ITER; ++i) dm[i] = 1e10f;
  if (tid == 0) { cidx[b * NP] = 0; slast = 0; }
  __syncthreads();
  for (int s = 1; s < NP; ++s) {
    int last = slast;
    float px = xyz[(b * N + last) * 3 + 0];
    float py = xyz[(b * N + last) * 3 + 1];
    float pz = xyz[(b * N + last) * 3 + 2];
    float bv = -1.f; int bi = 0;
#pragma unroll
    for (int i = 0; i < ITER; ++i) {
      int n = tid + i * 256;
      float dx = xyz[(b * N + n) * 3 + 0] - px;
      float dy = xyz[(b * N + n) * 3 + 1] - py;
      float dz = xyz[(b * N + n) * 3 + 2] - pz;
      float d = dx * dx + dy * dy + dz * dz;
      dm[i] = fminf(dm[i], d);
      if (dm[i] > bv) { bv = dm[i]; bi = n; }
    }
    sval[tid] = bv; sidx[tid] = bi;
    __syncthreads();
    for (int off = 128; off > 0; off >>= 1) {
      if (tid < off) {
        float v2 = sval[tid + off]; int i2 = sidx[tid + off];
        if (v2 > sval[tid] || (v2 == sval[tid] && i2 < sidx[tid])) { sval[tid] = v2; sidx[tid] = i2; }
      }
      __syncthreads();
    }
    if (tid == 0) { slast = sidx[0]; cidx[b * NP + s] = sidx[0]; }
    __syncthreads();
  }
  for (int s = tid; s < NP; s += 256) {
    int p = cidx[b * NP + s];
    centers[(b * NP + s) * 3 + 0] = xyz[(b * N + p) * 3 + 0];
    centers[(b * NP + s) * 3 + 1] = xyz[(b * N + p) * 3 + 1];
    centers[(b * NP + s) * 3 + 2] = xyz[(b * N + p) * 3 + 2];
  }
}

// ---------------- Ball query: one wave32 per center ----------------
template<int SRC_N, int S, int NS>
__global__ void bq_kernel(const float* __restrict__ centers, const float* __restrict__ xyz,
                          int* __restrict__ nidx, float r2) {
  const int lane = threadIdx.x & 31;
  const int w = blockIdx.x * 8 + (threadIdx.x >> 5);
  const int b = w / S, c = w % S;
  const float cx = centers[(b * S + c) * 3 + 0];
  const float cy = centers[(b * S + c) * 3 + 1];
  const float cz = centers[(b * S + c) * 3 + 2];
  int* out = nidx + (b * S + c) * NS;
  int count = 0, firstN = 0;
  bool haveFirst = false;
  for (int base = 0; base < SRC_N; base += 32) {
    int n = base + lane;
    float dx = xyz[(b * SRC_N + n) * 3 + 0] - cx;
    float dy = xyz[(b * SRC_N + n) * 3 + 1] - cy;
    float dz = xyz[(b * SRC_N + n) * 3 + 2] - cz;
    bool ok = (dx * dx + dy * dy + dz * dz) <= r2;
    unsigned long long m = __ballot(ok);
    unsigned mm = (unsigned)m;
    if (!haveFirst && mm) { firstN = base + __ffs(mm) - 1; haveFirst = true; }
    if (ok) {
      int pos = count + __popc(mm & ((1u << lane) - 1u));
      if (pos < NS) out[pos] = n;
    }
    count += __popc(mm);
    if (count >= NS) break;
  }
  for (int j = count + lane; j < NS; j += 32) out[j] = firstN;
}

// ---------------- Fused SA block (group + 3 MLP layers + maxpool) ----------------
// FT: dtype of gathered features; OT: dtype of pooled output.
template<int NS, int CF, int SRC_N, int S, int K0,
         int C0, int C1, int C2, int G, int ST0, int ST1, typename FT, typename OT>
__global__ void sa_kernel(const float* __restrict__ xyz, const FT* __restrict__ feats,
                          const float* __restrict__ centers, const int* __restrict__ nidx,
                          const _Float16* w0, const float* g0, const float* b0,
                          const _Float16* w1, const float* g1, const float* b1,
                          const _Float16* w2, const float* g2, const float* b2,
                          OT* __restrict__ outF) {
  constexpr int ROWS = G * NS; // 64
  __shared__ __align__(16) _Float16 buf0[ROWS * ST0];
  __shared__ __align__(16) _Float16 buf1[ROWS * ST1];
  const int b = blockIdx.y, tile = blockIdx.x, tid = threadIdx.x;
  const int cbase = tile * G;
  for (int idx = tid; idx < ROWS * K0; idx += 256) {
    int r = idx / K0, c = idx % K0;
    int ci = cbase + r / NS, j = r % NS;
    int p = nidx[(b * S + ci) * NS + j];
    float v = 0.f;
    if (c < 3)            v = xyz[(b * SRC_N + p) * 3 + c] - centers[(b * S + ci) * 3 + c];
    else if (c < 3 + CF)  v = (float)feats[(b * SRC_N + p) * CF + (c - 3)];
    buf0[r * ST0 + c] = (_Float16)v;
  }
  __syncthreads();
  layer_lds<ROWS>(buf0, ST0, K0, w0, g0, b0, buf1, ST1, C0);
  __syncthreads();
  layer_lds<ROWS>(buf1, ST1, C0, w1, g1, b1, buf0, ST0, C1);
  __syncthreads();
  layer_lds<ROWS>(buf0, ST0, C1, w2, g2, b2, buf1, ST1, C2);
  __syncthreads();
  for (int idx = tid; idx < G * C2; idx += 256) {
    int ci = idx / C2, ch = idx % C2;
    float m = (float)buf1[(ci * NS) * ST1 + ch];
    for (int j = 1; j < NS; ++j) m = fmaxf(m, (float)buf1[(ci * NS + j) * ST1 + ch]);
    outF[((b * S) + (cbase + ci)) * C2 + ch] = (OT)m;
  }
}

// ---------------- Global MLP 1: (2048x256 f16) x (512x256 f16)^T -> x1 (f16) ----------------
__global__ void g1_kernel(const _Float16* __restrict__ f2, const _Float16* __restrict__ W,
                          const float* __restrict__ gam, const float* __restrict__ bet,
                          _Float16* __restrict__ x1) {
  const int lane = threadIdx.x & 31, wv = threadIdx.x >> 5;
  const int lm = lane & 15, hi = lane >> 4;
  const int kbA = hi ? 8 : 0, kbB = hi ? 16 : 0;
  const int Mt = 128, Nt = 32;
  for (int t = blockIdx.x * 8 + wv; t < Mt * Nt; t += gridDim.x * 8) {
    int mt = t % Mt, nt = t / Mt;
    int row = mt * 16 + lm, ncol = nt * 16 + lm;
    v8f acc = tile_h(f2 + row * 256, W + ncol * 256, 256, kbA, kbB);
    float sc = gam[ncol] * INVS, bi = bet[ncol];
    int rb = mt * 16 + (hi ? 8 : 0);
#pragma unroll
    for (int r = 0; r < 8; ++r) {
      float v = acc[r] * sc + bi;
      x1[(rb + r) * 512 + ncol] = (_Float16)(v > 0.f ? v : 0.f);
    }
  }
}

// ---------------- Global MLP 2 (f16 in) fused with max-pool over S ----------------
__global__ void g2_kernel(const _Float16* __restrict__ x1, const _Float16* __restrict__ W,
                          const float* __restrict__ bias2, const float* __restrict__ gam,
                          const float* __restrict__ bet, int* __restrict__ pooledBits) {
  const int lane = threadIdx.x & 31, wv = threadIdx.x >> 5;
  const int lm = lane & 15, hi = lane >> 4;
  const int kbA = hi ? 8 : 0, kbB = hi ? 16 : 0;
  const int Mt = 128, Nt = 64;
  for (int t = blockIdx.x * 8 + wv; t < Mt * Nt; t += gridDim.x * 8) {
    int mt = t % Mt, nt = t / Mt;
    int row = mt * 16 + lm, ncol = nt * 16 + lm;
    v8f acc = tile_h(x1 + row * 512, W + ncol * 512, 512, kbA, kbB);
    float sc = gam[ncol] * INVS, bi = bet[ncol], bz = bias2[ncol];
    float m = 0.f;
#pragma unroll
    for (int r = 0; r < 8; ++r) {
      float v = (acc[r] + bz) * sc + bi;
      m = fmaxf(m, v > 0.f ? v : 0.f);
    }
    int bb = (mt * 16) / 128;                 // batch of this M-tile
    atomicMax(&pooledBits[bb * 1024 + ncol], __float_as_int(m));  // relu>=0 -> bit-monotonic
  }
}

__global__ void zero_kernel(int* p, int n) {
  int i = blockIdx.x * 256 + threadIdx.x;
  if (i < n) p[i] = 0;
}

// ---------------- Head: 1024->256 (bn+relu) -> 40, fp32 ----------------
__global__ void head_kernel(const float* __restrict__ pooled,
                            const float* __restrict__ hw1, const float* __restrict__ hb1,
                            const float* __restrict__ hg,  const float* __restrict__ hbe,
                            const float* __restrict__ hw2, const float* __restrict__ hb2,
                            float* __restrict__ out) {
  __shared__ float sh[16 * 256];
  const int tid = threadIdx.x;
  for (int o = tid; o < 16 * 256; o += 256) {
    int b = o / 256, ch = o % 256;
    const float* pr = pooled + b * 1024;
    const float* wr = hw1 + ch * 1024;
    float s = 0.f;
    for (int k = 0; k < 1024; ++k) s += pr[k] * wr[k];
    s += hb1[ch];
    s = s * hg[ch] * INVS + hbe[ch];
    sh[o] = s > 0.f ? s : 0.f;
  }
  __syncthreads();
  for (int o = tid; o < 16 * 40; o += 256) {
    int b = o / 40, c = o % 40;
    const float* hr = sh + b * 256;
    const float* wr = hw2 + c * 256;
    float s = 0.f;
    for (int k = 0; k < 256; ++k) s += hr[k] * wr[k];
    out[o] = s + hb2[c];
  }
}

// ---------------- Launch ----------------
extern "C" void kernel_launch(void* const* d_in, const int* in_sizes, int n_in,
                              void* d_out, int out_size, void* d_ws, size_t ws_size,
                              hipStream_t stream) {
  const float* xyz      = (const float*)d_in[0];
  const float* features = (const float*)d_in[1];
  const float *s1w0 = (const float*)d_in[2],  *s1g0 = (const float*)d_in[3],  *s1b0 = (const float*)d_in[4];
  const float *s1w1 = (const float*)d_in[5],  *s1g1 = (const float*)d_in[6],  *s1b1 = (const float*)d_in[7];
  const float *s1w2 = (const float*)d_in[8],  *s1g2 = (const float*)d_in[9],  *s1b2 = (const float*)d_in[10];
  const float *s2w0 = (const float*)d_in[11], *s2g0 = (const float*)d_in[12], *s2b0 = (const float*)d_in[13];
  const float *s2w1 = (const float*)d_in[14], *s2g1 = (const float*)d_in[15], *s2b1 = (const float*)d_in[16];
  const float *s2w2 = (const float*)d_in[17], *s2g2 = (const float*)d_in[18], *s2b2 = (const float*)d_in[19];
  const float *gw1 = (const float*)d_in[20], *gg1 = (const float*)d_in[21], *gb1 = (const float*)d_in[22];
  const float *gw2 = (const float*)d_in[23], *gbias2 = (const float*)d_in[24];
  const float *gg2 = (const float*)d_in[25], *gb2 = (const float*)d_in[26];
  const float *hw1 = (const float*)d_in[27], *hb1 = (const float*)d_in[28];
  const float *hg  = (const float*)d_in[29], *hbe = (const float*)d_in[30];
  const float *hw2 = (const float*)d_in[31], *hb2 = (const float*)d_in[32];
  float* out = (float*)d_out;

  char* ws = (char*)d_ws;
  size_t off = 0;
  auto alloc = [&](size_t bytes) { void* p = ws + off; off += (bytes + 255) & ~(size_t)255; return p; };
  int*       cidx1    = (int*)      alloc(16 * 512 * 4);
  float*     centers1 = (float*)    alloc(16 * 512 * 3 * 4);
  int*       nidx1    = (int*)      alloc((size_t)16 * 512 * 32 * 4);
  _Float16*  f1       = (_Float16*) alloc((size_t)16 * 512 * 128 * 2);
  int*       cidx2    = (int*)      alloc(16 * 128 * 4);
  float*     centers2 = (float*)    alloc(16 * 128 * 3 * 4);
  int*       nidx2    = (int*)      alloc((size_t)16 * 128 * 64 * 4);
  _Float16*  f2       = (_Float16*) alloc((size_t)16 * 128 * 256 * 2);
  _Float16*  x1       = (_Float16*) alloc((size_t)16 * 128 * 512 * 2);
  int*       pooled   = (int*)      alloc(16 * 1024 * 4);
  // pre-converted, zero-padded f16 weights
  _Float16*  w10h = (_Float16*) alloc((size_t)64  * 32  * 2);
  _Float16*  w11h = (_Float16*) alloc((size_t)64  * 64  * 2);
  _Float16*  w12h = (_Float16*) alloc((size_t)128 * 64  * 2);
  _Float16*  w20h = (_Float16*) alloc((size_t)128 * 160 * 2);
  _Float16*  w21h = (_Float16*) alloc((size_t)128 * 128 * 2);
  _Float16*  w22h = (_Float16*) alloc((size_t)256 * 128 * 2);
  _Float16*  gw1h = (_Float16*) alloc((size_t)512 * 256 * 2);
  _Float16*  gw2h = (_Float16*) alloc((size_t)1024 * 512 * 2);

  auto cvt = [&](const float* s, _Float16* d, int Cout, int Kreal, int Kpad) {
    int n = Cout * Kpad;
    cvtw_kernel<<<(n + 255) / 256, 256, 0, stream>>>(s, d, Cout, Kreal, Kpad);
  };
  cvt(s1w0, w10h, 64, 6, 32);
  cvt(s1w1, w11h, 64, 64, 64);
  cvt(s1w2, w12h, 128, 64, 64);
  cvt(s2w0, w20h, 128, 131, 160);
  cvt(s2w1, w21h, 128, 128, 128);
  cvt(s2w2, w22h, 256, 128, 128);
  cvt(gw1,  gw1h, 512, 256, 256);
  cvt(gw2,  gw2h, 1024, 512, 512);

  // SA1: N=4096 -> 512 centers, ns=32, 6->64->64->128
  fps_kernel<4096, 512, 16><<<16, 256, 0, stream>>>(xyz, cidx1, centers1);
  bq_kernel<4096, 512, 32><<<(16 * 512) / 8, 256, 0, stream>>>(centers1, xyz, nidx1, 0.2f * 0.2f);
  sa_kernel<32, 3, 4096, 512, 32, 64, 64, 128, 2, 72, 136, float, _Float16>
      <<<dim3(512 / 2, 16), 256, 0, stream>>>(xyz, features, centers1, nidx1,
                                              w10h, s1g0, s1b0, w11h, s1g1, s1b1, w12h, s1g2, s1b2, f1);
  // SA2: N=512 -> 128 centers, ns=64, 131->128->128->256
  fps_kernel<512, 128, 2><<<16, 256, 0, stream>>>(centers1, cidx2, centers2);
  bq_kernel<512, 128, 64><<<(16 * 128) / 8, 256, 0, stream>>>(centers2, centers1, nidx2, 0.4f * 0.4f);
  sa_kernel<64, 128, 512, 128, 160, 128, 128, 256, 1, 168, 264, _Float16, _Float16>
      <<<dim3(128, 16), 256, 0, stream>>>(centers1, f1, centers2, nidx2,
                                          w20h, s2g0, s2b0, w21h, s2g1, s2b1, w22h, s2g2, s2b2, f2);
  // Global MLPs + pooled max
  g1_kernel<<<64, 256, 0, stream>>>(f2, gw1h, gg1, gb1, x1);
  zero_kernel<<<(16 * 1024 + 255) / 256, 256, 0, stream>>>(pooled, 16 * 1024);
  g2_kernel<<<128, 256, 0, stream>>>(x1, gw2h, gbias2, gg2, gb2, pooled);
  // Head
  head_kernel<<<1, 256, 0, stream>>>((const float*)pooled, hw1, hb1, hg, hbe, hw2, hb2, out);
}
